// MultiHeadAttention_57277683860164
// MI455X (gfx1250) — matrix operand order repaired and checked by reading, not compiled, not run on hardware
//
#include <hip/hip_runtime.h>
#include <hip/hip_bf16.h>
#include <cstdint>

#define D_MODEL_ 1024
#define D_K_     64
#define SEQ_     2048
#define BH_      32            // B * H
#define BS_      4096          // B * S
#define HEAD_ELEMS_ 4194304    // B*H*S*D_K == B*S*D_MODEL

typedef __attribute__((ext_vector_type(16))) __bf16 v16bf;
typedef __attribute__((ext_vector_type(8)))  __bf16 v8bf;
typedef __attribute__((ext_vector_type(8)))  float  v8f;
typedef __attribute__((ext_vector_type(4)))  float  v4f;

static __device__ __forceinline__ v8f wmma_bf16(v16bf a, v16bf b, v8f c) {
  // 8 args: (neg_a, A, neg_b, B, c_mod, C, reuse_a, reuse_b)
  return __builtin_amdgcn_wmma_f32_16x16x32_bf16(false, a, false, b, (short)0, c,
                                                 false, false);
}

// ---- Fragment loaders -------------------------------------------------------
// A-matrix (16x32 bf16): lane L holds row M=L&15; contiguous K run of 16
// row[kb..kb+15], kb = k0 + (L>=16 ? 16 : 0).
static __device__ __forceinline__ v16bf load_a_f32(const float* __restrict__ row, int kb) {
  v4f f0 = *(const v4f*)(row + kb);
  v4f f1 = *(const v4f*)(row + kb + 4);
  v4f f2 = *(const v4f*)(row + kb + 8);
  v4f f3 = *(const v4f*)(row + kb + 12);
  v16bf r;
#pragma unroll
  for (int i = 0; i < 4; ++i) {
    r[i]      = (__bf16)f0[i];
    r[4 + i]  = (__bf16)f1[i];
    r[8 + i]  = (__bf16)f2[i];
    r[12 + i] = (__bf16)f3[i];
  }
  return r;
}

static __device__ __forceinline__ v16bf load_a_bf16(const __bf16* __restrict__ row, int kb) {
  v8bf h0 = *(const v8bf*)(row + kb);
  v8bf h1 = *(const v8bf*)(row + kb + 8);
  v16bf r;
#pragma unroll
  for (int i = 0; i < 8; ++i) { r[i] = h0[i]; r[8 + i] = h1[i]; }
  return r;
}

// B-matrix (32x16 bf16): lane L holds column N=L&15, contiguous K run of 16
// starting at k0 + (L>=16 ? 16 : 0).  `row` points at the memory row whose
// elements form that column (NT-gemm / row-of-W / row-of-K / row-of-Vt).
static __device__ __forceinline__ v16bf load_b_f32(const float* __restrict__ row, int kb) {
  v4f f0 = *(const v4f*)(row + kb);
  v4f f1 = *(const v4f*)(row + kb + 4);
  v4f f2 = *(const v4f*)(row + kb + 8);
  v4f f3 = *(const v4f*)(row + kb + 12);
  v16bf r;
#pragma unroll
  for (int i = 0; i < 4; ++i) {
    r[i]      = (__bf16)f0[i];
    r[4 + i]  = (__bf16)f1[i];
    r[8 + i]  = (__bf16)f2[i];
    r[12 + i] = (__bf16)f3[i];
  }
  return r;
}

static __device__ __forceinline__ v16bf load_b_bf16(const __bf16* __restrict__ row, int kb) {
  return *(const v16bf*)(row + kb);   // 32B, 32B-aligned
}

// ---- Kernel 1: fused Linear (x @ W^T + b) -> bf16 head-split layouts --------
// MODE 0: Qh[b,h,s,d]   MODE 1: Kh[b,h,s,d]   MODE 2: Vt[b,h,d,s] (transposed)
template <int MODE>
__global__ void mha_qkv_proj(const float* __restrict__ X, const float* __restrict__ W,
                             const float* __restrict__ bias, __bf16* __restrict__ out) {
  const int lane = threadIdx.x & 31;
  const int wave = threadIdx.x >> 5;
  const int hi   = lane >> 4;
  const int l15  = lane & 15;
  const int n0   = blockIdx.x * 16;                 // output col tile
  const int m0   = blockIdx.y * 128 + wave * 16;    // output row tile

  const float* arow = X + (size_t)(m0 + l15) * D_MODEL_;
  const int    n    = n0 + l15;
  const float* brow = W + (size_t)n * D_MODEL_;
  const int a_off = hi * 16, b_off = hi * 16;

  v8f c = {};
#pragma unroll 2
  for (int k0 = 0; k0 < D_MODEL_; k0 += 32) {
    if (k0 + 160 <= D_MODEL_) {
      __builtin_prefetch(arow + k0 + 128, 0, 0);
      __builtin_prefetch(brow + k0 + 128, 0, 0);
    }
    v16bf a = load_a_f32(arow, k0 + a_off);
    v16bf b = load_b_f32(brow, k0 + b_off);
    c = wmma_bf16(a, b, c);
  }

  const float bn = bias[n];
  const int h = n >> 6, d = n & 63;
  const int mbase = m0 + hi * 8;                // 8 consecutive rows per lane
  const int bb = mbase >> 11, sbase = mbase & (SEQ_ - 1);
  if (MODE == 2) {
    // Vt[(b*16+h)*64 + d][s]: fixed row per lane, 8 consecutive s -> one 16B store
    __bf16 tmp[8];
#pragma unroll
    for (int v = 0; v < 8; ++v) tmp[v] = (__bf16)(c[v] + bn);
    __bf16* dst = out + (((size_t)(bb * 16 + h) * D_K_ + d) * SEQ_ + sbase);
    *(v8bf*)dst = *(const v8bf*)tmp;
  } else {
#pragma unroll
    for (int v = 0; v < 8; ++v) {
      const int s = sbase + v;
      out[(((size_t)(bb * 16 + h) * SEQ_ + s) * D_K_) + d] = (__bf16)(c[v] + bn);
    }
  }
}

// ---- Kernel 2: scores (Q K^T * scale, mask) + softmax -> attn weights (f32) -
__global__ void mha_scores_softmax(const __bf16* __restrict__ Qh,
                                   const __bf16* __restrict__ Kh,
                                   const int* __restrict__ mask,
                                   float* __restrict__ attn) {
  constexpr int SROW = SEQ_ + 16;   // padded LDS row stride (bank-conflict relief)
  extern __shared__ float lds[];
  float* sc      = lds;                     // [16][SROW]
  float* red     = lds + 16 * SROW;         // [16][16]
  float* rowstat = red + 256;               // [16]

  const int bh   = blockIdx.x;              // (b*16 + h)
  const int q0   = blockIdx.y * 16;
  const int lane = threadIdx.x & 31;
  const int wave = threadIdx.x >> 5;
  const int hi   = lane >> 4, l15 = lane & 15;

  // A fragments: Q rows q0..q0+15, full K=64 (two 16x16x32 steps)
  const __bf16* qrow = Qh + ((size_t)bh * SEQ_ + (q0 + l15)) * D_K_;
  v16bf a0 = load_a_bf16(qrow, 0 + hi * 16);
  v16bf a1 = load_a_bf16(qrow, 32 + hi * 16);

  const float scale = 0.125f;  // 1/sqrt(64)

  for (int t = 0; t < 16; ++t) {
    const int n0 = wave * 256 + t * 16;
    const __bf16* krow = Kh + ((size_t)bh * SEQ_ + (n0 + l15)) * D_K_;
    v16bf b0 = load_b_bf16(krow, 0 + hi * 16);
    v16bf b1 = load_b_bf16(krow, 32 + hi * 16);
    v8f c = {};
    c = wmma_bf16(a0, b0, c);
    c = wmma_bf16(a1, b1, c);
    const int n = n0 + l15;
#pragma unroll
    for (int v = 0; v < 8; ++v) {
      const int r = hi * 8 + v;
      const int mval = mask[(size_t)(q0 + r) * SEQ_ + n];
      sc[r * SROW + n] = (mval == 0) ? -1e9f : c[v] * scale;
    }
  }
  __syncthreads();

  // row softmax: 16 threads cooperate per row
  const int r  = threadIdx.x >> 4;
  const int cp = threadIdx.x & 15;
  float mx = -3.0e38f;
  for (int cc = cp; cc < SEQ_; cc += 16) mx = fmaxf(mx, sc[r * SROW + cc]);
  red[r * 16 + cp] = mx;
  __syncthreads();
  mx = -3.0e38f;
#pragma unroll
  for (int i = 0; i < 16; ++i) mx = fmaxf(mx, red[r * 16 + i]);
  __syncthreads();

  float sum = 0.f;
  for (int cc = cp; cc < SEQ_; cc += 16) {
    float e = __expf(sc[r * SROW + cc] - mx);
    sc[r * SROW + cc] = e;
    sum += e;
  }
  red[r * 16 + cp] = sum;
  __syncthreads();
  sum = 0.f;
#pragma unroll
  for (int i = 0; i < 16; ++i) sum += red[r * 16 + i];
  if (cp == 0) rowstat[r] = 1.0f / sum;
  __syncthreads();

  // coalesced block-linear write of normalized weights
  float* dst = attn + ((size_t)bh * SEQ_ + q0) * SEQ_;
  for (int i = threadIdx.x; i < 16 * SEQ_; i += 256) {
    const int rr = i >> 11, cc = i & (SEQ_ - 1);
    dst[i] = sc[rr * SROW + cc] * rowstat[rr];
  }
}

// ---- Kernel 3: out_h = P @ V  -> concat-head bf16 [B*S, D_MODEL] ------------
__global__ void mha_pv(const float* __restrict__ P, const __bf16* __restrict__ Vt,
                       __bf16* __restrict__ outc) {
  const int bh   = blockIdx.x;
  const int lane = threadIdx.x & 31;
  const int wave = threadIdx.x >> 5;
  const int q0   = blockIdx.y * 128 + wave * 16;
  const int hi   = lane >> 4, l15 = lane & 15;
  const int bb   = bh >> 4, h = bh & 15;

  const float* prow = P + ((size_t)bh * SEQ_ + (q0 + l15)) * SEQ_;
  v8f c[4] = {};
  for (int k0 = 0; k0 < SEQ_; k0 += 32) {
    v16bf a = load_a_f32(prow, k0 + hi * 16);
#pragma unroll
    for (int t = 0; t < 4; ++t) {
      const int n = t * 16 + l15;  // d within head
      const __bf16* vrow = Vt + ((size_t)bh * D_K_ + n) * SEQ_;
      v16bf bm = load_b_bf16(vrow, k0 + hi * 16);
      c[t] = wmma_bf16(a, bm, c[t]);
    }
  }
#pragma unroll
  for (int t = 0; t < 4; ++t) {
    const int n = t * 16 + l15;
#pragma unroll
    for (int v = 0; v < 8; ++v) {
      const int q = q0 + hi * 8 + v;
      outc[((size_t)(bb * SEQ_ + q)) * D_MODEL_ + h * 64 + n] = (__bf16)c[t][v];
    }
  }
}

// ---- Kernel 4: final Linear (concat @ Wo^T + bo) -> fp32 output -------------
__global__ void mha_out_proj(const __bf16* __restrict__ A, const float* __restrict__ W,
                             const float* __restrict__ bias, float* __restrict__ out) {
  const int lane = threadIdx.x & 31;
  const int wave = threadIdx.x >> 5;
  const int hi   = lane >> 4, l15 = lane & 15;
  const int n0   = blockIdx.x * 16;
  const int m0   = blockIdx.y * 128 + wave * 16;

  const __bf16* arow = A + (size_t)(m0 + l15) * D_MODEL_;
  const int     n    = n0 + l15;
  const float*  brow = W + (size_t)n * D_MODEL_;

  v8f c = {};
#pragma unroll 2
  for (int k0 = 0; k0 < D_MODEL_; k0 += 32) {
    v16bf a = load_a_bf16(arow, k0 + hi * 16);
    v16bf b = load_b_f32(brow, k0 + hi * 16);
    c = wmma_bf16(a, b, c);
  }
  const float bn = bias[n];
#pragma unroll
  for (int v = 0; v < 8; ++v) {
    const int m = m0 + hi * 8 + v;
    out[(size_t)m * D_MODEL_ + n] = c[v] + bn;
  }
}

// ---- Host launcher ----------------------------------------------------------
extern "C" void kernel_launch(void* const* d_in, const int* in_sizes, int n_in,
                              void* d_out, int out_size, void* d_ws, size_t ws_size,
                              hipStream_t stream) {
  const float* q    = (const float*)d_in[0];
  const float* k    = (const float*)d_in[1];
  const float* v    = (const float*)d_in[2];
  const int*   mask = (const int*)d_in[3];
  const float* w_q  = (const float*)d_in[4];
  const float* b_q  = (const float*)d_in[5];
  const float* w_k  = (const float*)d_in[6];
  const float* b_k  = (const float*)d_in[7];
  const float* w_v  = (const float*)d_in[8];
  const float* b_v  = (const float*)d_in[9];
  const float* w_o  = (const float*)d_in[10];
  const float* b_o  = (const float*)d_in[11];

  float* outp = (float*)d_out;                 // [B,S,D_MODEL]
  float* attn = outp + (size_t)HEAD_ELEMS_;    // [B,H,S,S]

  __bf16* Qh   = (__bf16*)d_ws;
  __bf16* Kh   = Qh + HEAD_ELEMS_;
  __bf16* Vt   = Kh + HEAD_ELEMS_;
  __bf16* outc = Vt + HEAD_ELEMS_;

  const dim3 blk(256);
  const dim3 grid_proj(D_MODEL_ / 16, BS_ / 128);
  mha_qkv_proj<0><<<grid_proj, blk, 0, stream>>>(q, w_q, b_q, Qh);
  mha_qkv_proj<1><<<grid_proj, blk, 0, stream>>>(k, w_k, b_k, Kh);
  mha_qkv_proj<2><<<grid_proj, blk, 0, stream>>>(v, w_v, b_v, Vt);

  const size_t ldsBytes = (16 * (SEQ_ + 16) + 256 + 16) * sizeof(float); // ~133 KB
  mha_scores_softmax<<<dim3(BH_, SEQ_ / 16), blk, ldsBytes, stream>>>(Qh, Kh, mask, attn);

  mha_pv<<<dim3(BH_, SEQ_ / 128), blk, 0, stream>>>(attn, Vt, outc);

  mha_out_proj<<<grid_proj, blk, 0, stream>>>(outc, w_o, b_o, outp);
}
